// MambaLayer_66443144069654
// MI455X (gfx1250) — matrix operand order repaired and checked by reading, not compiled, not run on hardware
//
#include <hip/hip_runtime.h>
#include <hip/hip_bf16.h>
#include <math.h>

// ---- problem constants (from reference) ----
#define D_MODEL  128
#define D_STATE  16
#define D_CONV   4
#define D_INNER  256
#define DT_RANK  8
#define WIDTH    128
#define BATCH    8
#define SEQ      2048
#define BL       (BATCH * SEQ)        // 16384 token rows
#define LN_EPS   1e-5f

typedef __attribute__((ext_vector_type(16))) __bf16 v16bf;
typedef __attribute__((ext_vector_type(4)))  __bf16 bf16x4;
typedef __attribute__((ext_vector_type(8)))  float  v8f;

// =====================================================================
// Fused WMMA GEMM:  C[M,N] = act( A[M,K] @ W[N,K]^T + bias )
// Compile-time N,K; block = 256 threads = 8 wave32s.
// Tile = 32 rows x 128 cols: each wave owns one 16-col sub-tile and two
// 16x16 accumulators (rows 0-15 / 16-31) sharing one B fragment.
// DOUBLE-BUFFERED LDS: at step k we (a) issue the k+1 tile's global
// loads into registers, (b) ds_load fragments + 2 WMMAs from buf[k&1],
// (c) convert/store the prefetched tile into buf[(k+1)&1], (d) one
// barrier. Global loads overlap ds_load+WMMA; loads are batched (all 5
// issued before any convert) so multiple are in flight.
// ACT: 0 = none, 1 = ELU.
// =====================================================================
template <int ACT, bool HAS_BIAS, int N, int K>
__global__ __launch_bounds__(256)
void wmma_gemm_kernel(const float* __restrict__ A, const float* __restrict__ W,
                      const float* __restrict__ bias, float* __restrict__ C)
{
    // 40-element row stride (80 B) keeps 8/16-elem sub-rows 16B-aligned
    __shared__ __bf16 a_sm[2][32][40];
    __shared__ __bf16 w_sm[2][128][40];
    constexpr int NK = K / 32;

    const int tid  = threadIdx.x;
    const int wid  = tid >> 5;         // wave id 0..7 -> N sub-tile
    const int lane = tid & 31;
    const int half = lane >> 4;        // WMMA lane half
    const int l    = lane & 15;
    const int row0 = blockIdx.x * 32;
    const int n0   = blockIdx.y * 128;

    // staging coordinates (one float4 for A, four float4 for W)
    const int ar = tid >> 3;           // 0..31
    const int ac = (tid & 7) << 2;     // 0,4,...,28

    // batch ALL tile loads into registers before any convert/store
    auto load_tile = [&](int k0, float4& av, float4 (&wv)[4]) {
        av = *(const float4*)&A[(size_t)(row0 + ar) * K + k0 + ac];
        #pragma unroll
        for (int j = 0; j < 4; ++j) {
            int idx = tid + 256 * j;   // 0..1023
            int wr  = idx >> 3;        // 0..127
            int wc  = (idx & 7) << 2;
            int n   = n0 + wr;
            if ((N & 127) == 0 || n < N)          // folds away if N%128==0
                wv[j] = *(const float4*)&W[(size_t)n * K + k0 + wc];
            else
                wv[j] = make_float4(0.f, 0.f, 0.f, 0.f);
        }
    };
    auto store_tile = [&](int buf, const float4& av, const float4 (&wv)[4]) {
        *(bf16x4*)&a_sm[buf][ar][ac] =
            (bf16x4){ (__bf16)av.x, (__bf16)av.y, (__bf16)av.z, (__bf16)av.w };
        #pragma unroll
        for (int j = 0; j < 4; ++j) {
            int idx = tid + 256 * j;
            int wr  = idx >> 3;
            int wc  = (idx & 7) << 2;
            *(bf16x4*)&w_sm[buf][wr][wc] =
                (bf16x4){ (__bf16)wv[j].x, (__bf16)wv[j].y,
                          (__bf16)wv[j].z, (__bf16)wv[j].w };
        }
    };

    v8f acc0 = {}, acc1 = {};
    float4 av, wv[4];

    // prologue: stage tile 0 into buffer 0
    load_tile(0, av, wv);
    store_tile(0, av, wv);
    __syncthreads();

    #pragma unroll
    for (int k = 0; k < NK; ++k) {
        const int cur = k & 1;
        if (k + 1 < NK) load_tile((k + 1) * 32, av, wv);   // overlap w/ WMMA

        v16bf af0, af1, bfr;
        #pragma unroll
        for (int e = 0; e < 16; ++e) {
            // A 16x32 bf16 layout (ISA 7.12.2): contiguous 8-elem K groups
            int ka = (e & 7) + (half << 3) + ((e >> 3) << 4);
            af0[e] = a_sm[cur][l][ka];
            af1[e] = a_sm[cur][l + 16][ka];
            // B 32x16 layout: same K groups as A
            bfr[e] = w_sm[cur][(wid << 4) + l][ka];
        }
        acc0 = __builtin_amdgcn_wmma_f32_16x16x32_bf16(
                   false, af0, false, bfr, (short)0, acc0, false, false);
        acc1 = __builtin_amdgcn_wmma_f32_16x16x32_bf16(
                   false, af1, false, bfr, (short)0, acc1, false, false);

        if (k + 1 < NK) store_tile(cur ^ 1, av, wv);       // other buffer
        __syncthreads();
    }

    // C/D layout: element i -> row (i + 8*half), col = lane%16
    const int col = n0 + (wid << 4) + l;
    if ((N & 127) == 0 || col < N) {
        float b = HAS_BIAS ? bias[col] : 0.0f;
        #pragma unroll
        for (int i = 0; i < 8; ++i) {
            int r0 = row0 + i + (half << 3);
            float v0 = acc0[i] + b;
            float v1 = acc1[i] + b;
            if (ACT == 1) {
                v0 = (v0 > 0.0f) ? v0 : (__expf(v0) - 1.0f);   // ELU
                v1 = (v1 > 0.0f) ? v1 : (__expf(v1) - 1.0f);
            }
            C[(size_t)r0 * N + col]        = v0;
            C[(size_t)(r0 + 16) * N + col] = v1;
        }
    }
}

// =====================================================================
// Causal depthwise conv (width 4) + bias + SiLU.
// Reads the xh half of xz (stride 512), writes xh (BL x 256).
// =====================================================================
__global__ __launch_bounds__(256)
void conv_silu_kernel(const float* __restrict__ xz,
                      const float* __restrict__ conv_w,
                      const float* __restrict__ conv_b,
                      float* __restrict__ xh)
{
    int i = blockIdx.x * 256 + threadIdx.x;
    if (i >= BL * D_INNER) return;
    int e  = i & 255;
    int bl = i >> 8;
    int b  = bl / SEQ, t = bl % SEQ;
    float acc = conv_b[e];
    #pragma unroll
    for (int j = 0; j < D_CONV; ++j) {
        int tt = t - (D_CONV - 1) + j;
        if (tt >= 0)
            acc += conv_w[e * D_CONV + j] * xz[((size_t)(b * SEQ + tt)) * 512 + e];
    }
    xh[i] = acc / (1.0f + __expf(-acc));   // SiLU
}

// =====================================================================
// dt = softplus( proj[:, :8] @ dt_proj_w^T + dt_proj_b )   (K=8, tiny)
// =====================================================================
__global__ __launch_bounds__(256)
void dtproj_kernel(const float* __restrict__ proj,
                   const float* __restrict__ dtw,
                   const float* __restrict__ dtb,
                   float* __restrict__ dt)
{
    int i = blockIdx.x * 256 + threadIdx.x;
    if (i >= BL * D_INNER) return;
    int e  = i & 255;
    int bl = i >> 8;
    float acc = dtb[e];
    #pragma unroll
    for (int r = 0; r < DT_RANK; ++r)
        acc += proj[(size_t)bl * 40 + r] * dtw[e * DT_RANK + r];
    dt[i] = (acc > 20.0f) ? acc : log1pf(__expf(acc));   // softplus
}

// =====================================================================
// Selective scan, fused with D-skip and SiLU(z) gating.
// One thread per (b, d, s): 32768 threads = 1024 wave32s.
// Software-pipelined: step t+1's operands are loaded before step t's
// exp/shfl dependency chain; prefetch hints pull L2-resident lines
// toward the WGP ahead of the serial recurrence (the layer's critical
// path).
// =====================================================================
__global__ __launch_bounds__(256)
void scan_kernel(const float* __restrict__ dt, const float* __restrict__ xh,
                 const float* __restrict__ proj, const float* __restrict__ xz,
                 const float* __restrict__ A_log, const float* __restrict__ Dp,
                 float* __restrict__ y)
{
    int b      = blockIdx.x >> 4;        // batch 0..7
    int dchunk = blockIdx.x & 15;        // channel chunk 0..15
    int s      = threadIdx.x & 15;       // state index
    int d      = dchunk * 16 + (threadIdx.x >> 4);

    float Ads = -__expf(A_log[d * D_STATE + s]);   // A = -exp(A_log)
    float Dd  = Dp[d];
    float h   = 0.0f;
    const size_t base = (size_t)b * SEQ;

    // preload t = 0
    float dtv = dt[base * 256 + d];
    float xv  = xh[base * 256 + d];
    float Bt  = proj[base * 40 + DT_RANK + s];
    float Ct  = proj[base * 40 + DT_RANK + D_STATE + s];

    for (int t = 0; t < SEQ; ++t) {
        // issue next step's loads before this step's dependency chain
        float dtn = 0.0f, xn = 0.0f, Bn = 0.0f, Cn = 0.0f;
        if (t + 1 < SEQ) {
            size_t nidx = base + t + 1;
            dtn = dt[nidx * 256 + d];
            xn  = xh[nidx * 256 + d];
            Bn  = proj[nidx * 40 + DT_RANK + s];
            Cn  = proj[nidx * 40 + DT_RANK + D_STATE + s];
        }
        if (t + 16 < SEQ) {   // stream-ahead hints (global_prefetch_b8)
            __builtin_prefetch(&dt[(base + t + 16) * 256 + d], 0, 3);
            __builtin_prefetch(&proj[(base + t + 16) * 40 + s], 0, 3);
        }

        float dA = __expf(dtv * Ads);
        h = dA * h + (dtv * xv) * Bt;
        float p = h * Ct;
        // reduce over the 16 states (xor of bits 0..3 stays in-group)
        p += __shfl_xor(p, 8, 32);
        p += __shfl_xor(p, 4, 32);
        p += __shfl_xor(p, 2, 32);
        p += __shfl_xor(p, 1, 32);
        if (s == 0) {
            size_t idx = base + t;
            float zv = xz[idx * 512 + 256 + d];          // z half of xz
            float g  = zv / (1.0f + __expf(-zv));        // SiLU(z)
            y[idx * 256 + d] = (p + xv * Dd) * g;
        }
        dtv = dtn; xv = xn; Bt = Bn; Ct = Cn;
    }
}

// =====================================================================
// r = h + m;  LayerNorm over 128;  zero rows where mask is set.
// One 128-thread block per token row.
// =====================================================================
__global__ __launch_bounds__(128)
void ln_kernel(const float* __restrict__ h, const float* __restrict__ m,
               const unsigned char* __restrict__ mask,
               const float* __restrict__ g, const float* __restrict__ be,
               float* __restrict__ out)
{
    __shared__ float red[128];
    int row = blockIdx.x, d = threadIdx.x;
    float r = h[(size_t)row * 128 + d] + m[(size_t)row * 128 + d];

    red[d] = r; __syncthreads();
    for (int off = 64; off > 0; off >>= 1) {
        if (d < off) red[d] += red[d + off];
        __syncthreads();
    }
    float mu = red[0] * (1.0f / 128.0f);
    __syncthreads();
    float c = r - mu;
    red[d] = c * c; __syncthreads();
    for (int off = 64; off > 0; off >>= 1) {
        if (d < off) red[d] += red[d + off];
        __syncthreads();
    }
    float var = red[0] * (1.0f / 128.0f);
    float o = c * rsqrtf(var + LN_EPS) * g[d] + be[d];
    o *= mask[row] ? 0.0f : 1.0f;                  // out * (~mask)
    out[(size_t)row * 128 + d] = o;
}

// =====================================================================
extern "C" void kernel_launch(void* const* d_in, const int* in_sizes, int n_in,
                              void* d_out, int out_size, void* d_ws, size_t ws_size,
                              hipStream_t stream)
{
    (void)in_sizes; (void)n_in; (void)out_size; (void)ws_size;
    const float*         x         = (const float*)d_in[0];
    const unsigned char* mask      = (const unsigned char*)d_in[1];  // bool mask
    const float*         in_proj_w = (const float*)d_in[2];
    const float*         conv_w    = (const float*)d_in[3];
    const float*         conv_b    = (const float*)d_in[4];
    const float*         x_proj_w  = (const float*)d_in[5];
    const float*         dt_proj_w = (const float*)d_in[6];
    const float*         dt_proj_b = (const float*)d_in[7];
    const float*         A_log     = (const float*)d_in[8];
    const float*         Dp        = (const float*)d_in[9];
    const float*         out_proj_w= (const float*)d_in[10];
    const float*         ln_g      = (const float*)d_in[11];
    const float*         ln_b      = (const float*)d_in[12];
    const float*         w1        = (const float*)d_in[13];
    const float*         b1        = (const float*)d_in[14];
    const float*         w2        = (const float*)d_in[15];
    const float*         b2        = (const float*)d_in[16];

    // workspace carve-out (floats per token row, cumulative)
    char* ws = (char*)d_ws;
    float* xz   = (float*)(ws);                                  // BL x 512
    float* xh   = (float*)(ws + (size_t)BL * 512 * 4);           // BL x 256
    float* proj = (float*)(ws + (size_t)BL * 768 * 4);           // BL x 40
    float* dt   = (float*)(ws + (size_t)BL * 808 * 4);           // BL x 256
    float* ybuf = (float*)(ws + (size_t)BL * 1064 * 4);          // BL x 256
    float* hbuf = (float*)(ws + (size_t)BL * 1320 * 4);          // BL x 128
    float* m1   = (float*)(ws + (size_t)BL * 1448 * 4);          // BL x 128
    float* m2   = (float*)(ws + (size_t)BL * 1576 * 4);          // BL x 128

    dim3 blk(256);

    // 1) xz = x @ in_proj_w^T                 (M=16384, N=512, K=128)
    wmma_gemm_kernel<0, false, 512, 128><<<dim3(BL / 32, 4), blk, 0, stream>>>(
        x, in_proj_w, nullptr, xz);

    // 2) xh = silu(depthwise_conv(xz[:, :256]) + conv_b)
    conv_silu_kernel<<<(BL * D_INNER) / 256, blk, 0, stream>>>(xz, conv_w, conv_b, xh);

    // 3) proj = xh @ x_proj_w^T               (N=40, column-guarded)
    wmma_gemm_kernel<0, false, 40, 256><<<dim3(BL / 32, 1), blk, 0, stream>>>(
        xh, x_proj_w, nullptr, proj);

    // 4) dt = softplus(proj[:, :8] @ dt_proj_w^T + dt_proj_b)
    dtproj_kernel<<<(BL * D_INNER) / 256, blk, 0, stream>>>(proj, dt_proj_w, dt_proj_b, dt);

    // 5) selective scan + D skip + SiLU(z) gate  ->  ybuf
    scan_kernel<<<BATCH * 16, blk, 0, stream>>>(dt, xh, proj, xz, A_log, Dp, ybuf);

    // 6) hbuf = ybuf @ out_proj_w^T           (N=128, K=256)
    wmma_gemm_kernel<0, false, 128, 256><<<dim3(BL / 32, 1), blk, 0, stream>>>(
        ybuf, out_proj_w, nullptr, hbuf);

    // 7) m1 = elu(hbuf @ w1^T + b1)
    wmma_gemm_kernel<1, true, 128, 128><<<dim3(BL / 32, 1), blk, 0, stream>>>(
        hbuf, w1, b1, m1);

    // 8) m2 = elu(m1 @ w2^T + b2)
    wmma_gemm_kernel<1, true, 128, 128><<<dim3(BL / 32, 1), blk, 0, stream>>>(
        m1, w2, b2, m2);

    // 9) LayerNorm(hbuf + m2) * ln_g + ln_b, masked
    ln_kernel<<<BL, 128, 0, stream>>>(hbuf, m2, mask, ln_g, ln_b, (float*)d_out);
}
